// DeformableResBlock_66907000537869
// MI455X (gfx1250) — matrix operand order repaired and checked by reading, not compiled, not run on hardware
//
#include <hip/hip_runtime.h>

// ---------------------------------------------------------------------------
// DeformableResBlock for MI455X (gfx1250, wave32, WMMA)
// deform convs -> implicit GEMM on v_wmma_f32_16x16x32_bf16 with bf16x3 split
// ---------------------------------------------------------------------------

#define Bn   4
#define CIN  64
#define COUT 128
#define Hh   128
#define Ww   128
#define HW   (Hh * Ww)
#define NPIX (Bn * HW)      // 65536
#define EPS  1e-5f

typedef __attribute__((ext_vector_type(16))) __bf16          v16bf;
typedef __attribute__((ext_vector_type(16))) unsigned short  v16us;
typedef __attribute__((ext_vector_type(8)))  unsigned short  v8us;
typedef __attribute__((ext_vector_type(8)))  float           v8f;

__device__ __forceinline__ unsigned short f32_to_bf16(float f) {
    unsigned int u = __float_as_uint(f);
    unsigned int r = u + 0x7FFFu + ((u >> 16) & 1u);   // round-to-nearest-even
    return (unsigned short)(r >> 16);
}
__device__ __forceinline__ float bf16_to_f32(unsigned short h) {
    return __uint_as_float(((unsigned int)h) << 16);
}

__device__ __forceinline__ v16bf make_frag(v8us a, v8us b) {
    union { v16us us; v16bf bf; } u;
#pragma unroll
    for (int e = 0; e < 8; ++e) { u.us[e] = a[e]; u.us[e + 8] = b[e]; }
    return u.bf;
}

__device__ __forceinline__ v8f wmma_bf16(v16bf a, v16bf b, v8f c) {
    // (neg_a, A, neg_b, B, c_mod, C, reuse_a, reuse_b)
    return __builtin_amdgcn_wmma_f32_16x16x32_bf16(false, a, false, b,
                                                   (short)0, c, false, false);
}

// ---------------------------------------------------------------------------
// BN fold: scale = g / sqrt(v+eps), shift = b - m*scale   (3 BN sets, 128 ch)
// ---------------------------------------------------------------------------
__global__ void bn_fold_kernel(const float* g1, const float* b1, const float* m1, const float* v1,
                               const float* g2, const float* b2, const float* m2, const float* v2,
                               const float* gs, const float* bs, const float* ms, const float* vs,
                               float* sc1, float* sh1, float* sc2, float* sh2,
                               float* scS, float* shS) {
    int c = threadIdx.x;
    if (c < COUT) {
        float s;
        s = g1[c] * __frsqrt_rn(v1[c] + EPS); sc1[c] = s; sh1[c] = b1[c] - m1[c] * s;
        s = g2[c] * __frsqrt_rn(v2[c] + EPS); sc2[c] = s; sh2[c] = b2[c] - m2[c] * s;
        s = gs[c] * __frsqrt_rn(vs[c] + EPS); scS[c] = s; shS[c] = bs[c] - ms[c] * s;
    }
}

// ---------------------------------------------------------------------------
// Pack deform weights [COUT, Cin, 9] (f32) into per-lane WMMA B-fragment order,
// split into bf16 hi / lo.  B order used here (contiguous K, A matches): lane half h, elem e ->
// K = 16*h + e within the 32-K block; N = lane & 15.
// packed[(((nb*KB + kb)*32 + lane)*16 + e)]
// ---------------------------------------------------------------------------
__global__ void pack_b_kernel(const float* __restrict__ w, int Cin, int KB,
                              unsigned short* __restrict__ hi,
                              unsigned short* __restrict__ lo) {
    int idx = blockIdx.x * 256 + threadIdx.x;
    int total = KB * 4096;                        // 8 nb * KB * 32 * 16
    if (idx >= total) return;
    int e    = idx & 15;
    int lane = (idx >> 4) & 31;
    int kb   = (idx >> 9) % KB;
    int nb   = (idx >> 9) / KB;
    int half = lane >> 4;
    int n    = lane & 15;
    int kdim = kb * 32 + 16 * half + e;           // global K index (c*9 + tap)
    int c    = kdim / 9;
    int kt   = kdim - c * 9;
    float f  = w[((size_t)(nb * 16 + n) * Cin + c) * 9 + kt];
    unsigned short h = f32_to_bf16(f);
    hi[idx] = h;
    lo[idx] = f32_to_bf16(f - bf16_to_f32(h));
}

// ---------------------------------------------------------------------------
// identity = bn_s(conv1x1(x, sw))  : thread per (pixel, cout)
// ---------------------------------------------------------------------------
__global__ void identity_kernel(const float* __restrict__ x,
                                const float* __restrict__ sw,
                                const float* __restrict__ scS,
                                const float* __restrict__ shS,
                                float* __restrict__ ident) {
    long idx = (long)blockIdx.x * 256 + threadIdx.x;   // NPIX * 128
    int co = (int)(idx & 127);
    int p  = (int)(idx >> 7);
    int b  = p / HW;
    int rr = p - b * HW;
    const float* xp = x + (size_t)b * CIN * HW + rr;
    const float* wp = sw + (size_t)co * CIN;
    float acc = 0.f;
#pragma unroll 8
    for (int c = 0; c < CIN; ++c) acc = fmaf(xp[(size_t)c * HW], wp[c], acc);
    ident[((size_t)b * COUT + co) * HW + rr] = acc * scS[co] + shS[co];
}

// ---------------------------------------------------------------------------
// off = conv3x3(src, wgt) + bias : thread per (pixel, oc), 18 output channels
// ---------------------------------------------------------------------------
__global__ void offset_conv_kernel(const float* __restrict__ src, int Cin,
                                   const float* __restrict__ wgt,
                                   const float* __restrict__ bias,
                                   float* __restrict__ off) {
    long idx = (long)blockIdx.x * 256 + threadIdx.x;   // NPIX * 18
    int p  = (int)(idx / 18);
    int oc = (int)(idx - (long)p * 18);
    int b  = p / HW;
    int rr = p - b * HW;
    int y  = rr >> 7;
    int x  = rr & 127;
    float acc = bias[oc];
    for (int c = 0; c < Cin; ++c) {
        const float* plane = src + ((size_t)b * Cin + c) * HW;
        const float* wp    = wgt + ((size_t)oc * Cin + c) * 9;
#pragma unroll
        for (int ii = 0; ii < 3; ++ii) {
            int yy = y + ii - 1;
            if (yy < 0 || yy >= Hh) continue;
#pragma unroll
            for (int jj = 0; jj < 3; ++jj) {
                int xx = x + jj - 1;
                if (xx < 0 || xx >= Ww) continue;
                acc = fmaf(plane[yy * Ww + xx], wp[ii * 3 + jj], acc);
            }
        }
    }
    off[((size_t)b * 18 + oc) * HW + rr] = acc;
}

// ---------------------------------------------------------------------------
// Deformable conv as implicit GEMM on WMMA bf16 (bf16x3 split for ~fp32).
//   tile: 16 pixels (M) x 128 couts (N);  8 waves, wave w owns N-block w.
//   K is consumed in 576-wide chunks (64 channels x 9 taps).
//   epilogue: BN, optional identity add, ReLU.
// ---------------------------------------------------------------------------
__global__ __launch_bounds__(256)
void deform_gemm_kernel(const float* __restrict__ src, int Cin, int nChunks, int KBtot,
                        const float* __restrict__ off,
                        const unsigned short* __restrict__ pBhi,
                        const unsigned short* __restrict__ pBlo,
                        const float* __restrict__ scale,
                        const float* __restrict__ shift,
                        const float* __restrict__ ident,   // nullptr for layer 1
                        float* __restrict__ dst) {
    alignas(16) __shared__ unsigned short sAhi[16 * 576];
    alignas(16) __shared__ unsigned short sAlo[16 * 576];
    __shared__ int   sIdx[144 * 4];
    __shared__ float sWt [144 * 4];

    const int tid = threadIdx.x;
    const int p0  = blockIdx.x * 16;

    // ---- phase 1: bilinear sample metadata for 16 pixels x 9 taps ----
    if (tid < 144) {
        int i = tid / 9, k = tid - i * 9;
        int p = p0 + i;
        int b = p / HW;
        int rr = p - b * HW;
        int y = rr >> 7, x = rr & 127;
        float dy = off[((size_t)b * 18 + 2 * k) * HW + rr];
        float dx = off[((size_t)b * 18 + 2 * k + 1) * HW + rr];
        float ys = (float)(y + (k / 3) - 1) + dy;
        float xs = (float)(x + (k % 3) - 1) + dx;
        float y0f = floorf(ys), x0f = floorf(xs);
        float fy = ys - y0f, fx = xs - x0f;
        int iy0 = (int)y0f, ix0 = (int)x0f;
        float w4[4] = { (1.f - fy) * (1.f - fx), (1.f - fy) * fx,
                        fy * (1.f - fx),         fy * fx };
        int yq[4] = { iy0, iy0, iy0 + 1, iy0 + 1 };
        int xq[4] = { ix0, ix0 + 1, ix0, ix0 + 1 };
        int t4 = tid * 4;
#pragma unroll
        for (int q = 0; q < 4; ++q) {
            bool valid = (yq[q] >= 0) && (yq[q] < Hh) && (xq[q] >= 0) && (xq[q] < Ww);
            int yc = min(max(yq[q], 0), Hh - 1);
            int xc = min(max(xq[q], 0), Ww - 1);
            sIdx[t4 + q] = yc * Ww + xc;
            sWt [t4 + q] = valid ? w4[q] : 0.f;
        }
    }
    __syncthreads();

    const int lane = tid & 31;
    const int half = lane >> 4;
    const int n    = lane & 15;          // column within N-block; also A row id
    const int nb   = tid >> 5;           // wave id == N-block (8 x 16 = 128)

    v8f acc = { 0.f, 0.f, 0.f, 0.f, 0.f, 0.f, 0.f, 0.f };

    for (int ch = 0; ch < nChunks; ++ch) {
        // ---- phase 2: build 16 x 576 bf16 hi/lo im2col patch in LDS ----
        for (int j = tid; j < 144 * 64; j += 256) {
            int cl = j / 144;
            int r2 = j - cl * 144;                 // i*9 + k
            int i  = r2 / 9;
            int k  = r2 - i * 9;
            int p  = p0 + i;
            int b  = p / HW;
            int c  = ch * 64 + cl;
            const float* plane = src + ((size_t)b * Cin + c) * HW;
            int t4 = r2 * 4;
            float v = sWt[t4]     * plane[sIdx[t4]]
                    + sWt[t4 + 1] * plane[sIdx[t4 + 1]]
                    + sWt[t4 + 2] * plane[sIdx[t4 + 2]]
                    + sWt[t4 + 3] * plane[sIdx[t4 + 3]];
            unsigned short hi = f32_to_bf16(v);
            int addr = i * 576 + cl * 9 + k;
            sAhi[addr] = hi;
            sAlo[addr] = f32_to_bf16(v - bf16_to_f32(hi));
        }
        __syncthreads();

        // ---- phase 3: 18 WMMA K-steps over this chunk ----
        for (int kb = 0; kb < 18; ++kb) {
            // A fragment: per-lane contiguous K = 16*half + e (matches packed B)
            int aoff = n * 576 + kb * 32 + 16 * half;
            v8us ah1 = *(const v8us*)(sAhi + aoff);
            v8us ah2 = *(const v8us*)(sAhi + aoff + 8);
            v8us al1 = *(const v8us*)(sAlo + aoff);
            v8us al2 = *(const v8us*)(sAlo + aoff + 8);
            // B fragment: prepacked, 32 contiguous bytes per lane
            int kbg = ch * 18 + kb;
            size_t boff = (((size_t)nb * KBtot + kbg) * 32 + lane) * 16;
            v8us bh1 = *(const v8us*)(pBhi + boff);
            v8us bh2 = *(const v8us*)(pBhi + boff + 8);
            v8us bl1 = *(const v8us*)(pBlo + boff);
            v8us bl2 = *(const v8us*)(pBlo + boff + 8);

            v16bf aHi = make_frag(ah1, ah2);
            v16bf aLo = make_frag(al1, al2);
            v16bf bHi = make_frag(bh1, bh2);
            v16bf bLo = make_frag(bl1, bl2);

            acc = wmma_bf16(aHi, bHi, acc);   // hi*hi
            acc = wmma_bf16(aHi, bLo, acc);   // hi*lo
            acc = wmma_bf16(aLo, bHi, acc);   // lo*hi
        }
        __syncthreads();
    }

    // ---- epilogue: BN + (identity) + ReLU ----
    // C/D layout: vgpr v -> M = v + 8*half, N = lane & 15
#pragma unroll
    for (int v = 0; v < 8; ++v) {
        int m  = v + 8 * half;
        int p  = p0 + m;
        int b  = p / HW;
        int rr = p - b * HW;
        int co = nb * 16 + n;
        float val = acc[v] * scale[co] + shift[co];
        size_t oidx = ((size_t)b * COUT + co) * HW + rr;
        if (ident) val += ident[oidx];
        dst[oidx] = fmaxf(val, 0.f);
    }
}

// ---------------------------------------------------------------------------
// host side
// ---------------------------------------------------------------------------
extern "C" void kernel_launch(void* const* d_in, const int* in_sizes, int n_in,
                              void* d_out, int out_size, void* d_ws, size_t ws_size,
                              hipStream_t stream) {
    const float* x   = (const float*)d_in[0];
    const float* ow1 = (const float*)d_in[1];
    const float* ob1 = (const float*)d_in[2];
    const float* dw1 = (const float*)d_in[3];
    const float* g1  = (const float*)d_in[4];
    const float* b1  = (const float*)d_in[5];
    const float* m1  = (const float*)d_in[6];
    const float* v1  = (const float*)d_in[7];
    const float* ow2 = (const float*)d_in[8];
    const float* ob2 = (const float*)d_in[9];
    const float* dw2 = (const float*)d_in[10];
    const float* g2  = (const float*)d_in[11];
    const float* b2  = (const float*)d_in[12];
    const float* m2  = (const float*)d_in[13];
    const float* v2  = (const float*)d_in[14];
    const float* sw  = (const float*)d_in[15];
    const float* gs  = (const float*)d_in[16];
    const float* bs  = (const float*)d_in[17];
    const float* ms  = (const float*)d_in[18];
    const float* vs  = (const float*)d_in[19];
    float* out = (float*)d_out;

    // workspace carve-up (~73 MB)
    char*  ws = (char*)d_ws;
    size_t o  = 0;
    auto carve = [&](size_t bytes) -> char* {
        char* p = ws + o;
        o = (o + bytes + 255) & ~(size_t)255;
        return p;
    };
    float* offb  = (float*)carve((size_t)Bn * 18 * HW * sizeof(float));     // 4.7 MB
    float* ident = (float*)carve((size_t)Bn * COUT * HW * sizeof(float));   // 33.5 MB
    float* out1  = (float*)carve((size_t)Bn * COUT * HW * sizeof(float));   // 33.5 MB
    float* sc1   = (float*)carve(COUT * sizeof(float));
    float* sh1   = (float*)carve(COUT * sizeof(float));
    float* sc2   = (float*)carve(COUT * sizeof(float));
    float* sh2   = (float*)carve(COUT * sizeof(float));
    float* scS   = (float*)carve(COUT * sizeof(float));
    float* shS   = (float*)carve(COUT * sizeof(float));
    unsigned short* pB1h = (unsigned short*)carve((size_t)18 * 4096 * 2);   // 147 KB
    unsigned short* pB1l = (unsigned short*)carve((size_t)18 * 4096 * 2);
    unsigned short* pB2h = (unsigned short*)carve((size_t)36 * 4096 * 2);   // 295 KB
    unsigned short* pB2l = (unsigned short*)carve((size_t)36 * 4096 * 2);

    // 1) fold BN params
    bn_fold_kernel<<<1, 128, 0, stream>>>(g1, b1, m1, v1, g2, b2, m2, v2,
                                          gs, bs, ms, vs,
                                          sc1, sh1, sc2, sh2, scS, shS);
    // 2) pack deform weights to WMMA B-fragment order (bf16 hi/lo)
    pack_b_kernel<<<(18 * 4096 + 255) / 256, 256, 0, stream>>>(dw1, CIN, 18, pB1h, pB1l);
    pack_b_kernel<<<(36 * 4096 + 255) / 256, 256, 0, stream>>>(dw2, COUT, 36, pB2h, pB2l);
    // 3) identity branch: conv1x1 + BN_s
    identity_kernel<<<(NPIX * 128) / 256, 256, 0, stream>>>(x, sw, scS, shS, ident);
    // 4) offset conv 1 (x -> 18 ch)
    offset_conv_kernel<<<(NPIX * 18) / 256, 256, 0, stream>>>(x, CIN, ow1, ob1, offb);
    // 5) deform conv 1 (+BN1+ReLU) via WMMA implicit GEMM -> out1
    deform_gemm_kernel<<<NPIX / 16, 256, 0, stream>>>(x, CIN, 1, 18, offb,
                                                      pB1h, pB1l, sc1, sh1,
                                                      nullptr, out1);
    // 6) offset conv 2 (out1 -> 18 ch)
    offset_conv_kernel<<<(NPIX * 18) / 256, 256, 0, stream>>>(out1, COUT, ow2, ob2, offb);
    // 7) deform conv 2 (+BN2 +identity +ReLU) via WMMA implicit GEMM -> d_out
    deform_gemm_kernel<<<NPIX / 16, 256, 0, stream>>>(out1, COUT, 2, 36, offb,
                                                      pB2h, pB2l, sc2, sh2,
                                                      ident, out);
}
